// EncoderLayer_15728170238259
// MI455X (gfx1250) — matrix operand order repaired and checked by reading, not compiled, not run on hardware
//
#include <hip/hip_runtime.h>
#include <hip/hip_bf16.h>

// ---------------------------------------------------------------------------
// Types for CDNA5 WMMA (wave32): D(16x16 f32) = A(16x32 bf16) * B(32x16 bf16) + C
// ---------------------------------------------------------------------------
typedef __attribute__((ext_vector_type(16))) __bf16 v16bf;
typedef __attribute__((ext_vector_type(8)))  __bf16 v8bf;
typedef __attribute__((ext_vector_type(8)))  float  v8f;
typedef unsigned int u32x4 __attribute__((ext_vector_type(4)));
typedef int          i32x4 __attribute__((ext_vector_type(4)));
typedef int          i32x8 __attribute__((ext_vector_type(8)));

#define WMMA_BF16(a, b, c) \
  __builtin_amdgcn_wmma_f32_16x16x32_bf16(false, (a), false, (b), (short)0, (c), false, false)

#define NEGV (-4294967295.0f)   // tf padding value -(2^32)+1
#define GST 40                  // LDS tile row stride in halves (16B-aligned rows)
#define TILE_HALVES (128 * GST) // one 128x32 tile with padding

#if defined(__gfx1250__) && __has_builtin(__builtin_amdgcn_tensor_load_to_lds) && \
    __has_builtin(__builtin_amdgcn_s_wait_tensorcnt)
#define USE_TDM 1
#else
#define USE_TDM 0
#endif

#if __has_include(<hip/amd_detail/amd_gfx1250_TDM.h>)
#define TDM_SIX_ARGS 1
#else
#define TDM_SIX_ARGS 0
#endif

// Load a 16-element bf16 fragment as two 16-byte chunks (A or B frag layouts).
static __device__ inline v16bf ld16x2(const __bf16* p0, const __bf16* p1) {
  v8bf lo = *(const v8bf*)p0;
  v8bf hi = *(const v8bf*)p1;
  v16bf r;
#pragma unroll
  for (int i = 0; i < 8; ++i) { r[i] = lo[i]; r[i + 8] = hi[i]; }
  return r;
}

// ---------------------------------------------------------------------------
// TDM: issue an async 2D tile load (tile_d0 x tile_d1 bf16) global -> LDS,
// with LDS padding of 4 DWORDs after every 16 DWORDs (row 32 bf16 -> stride 40).
// Descriptor packing per CDNA5 ISA section 8.3/8.4 (D# groups 0/1; 2D tensor).
// ---------------------------------------------------------------------------
static __device__ inline void tdm_load_tile_2d(unsigned lds_byte_addr,
                                               const __bf16* gsrc,
                                               unsigned tensor_d0, unsigned tensor_d1,
                                               unsigned tile_d0, unsigned tile_d1,
                                               unsigned stride0_elems) {
#if USE_TDM
  unsigned long long ga = (unsigned long long)(uintptr_t)gsrc;
  u32x4 g0;
  g0[0] = 1u;                                   // count=1 (valid user descriptor)
  g0[1] = lds_byte_addr;                        // LDS dest byte address
  g0[2] = (unsigned)ga;                         // global_addr[31:0]
  g0[3] = (unsigned)((ga >> 32) & 0x01FFFFFFu)  // global_addr[56:32]
          | (2u << 30);                         // type=2 ("image")
  i32x8 g1;
  g1[0] = (int)((1u << 16)      // data_size = 2 bytes
              | (1u << 20)      // pad_enable
              | (3u << 22)      // pad_interval: 16 DWORDs
              | (3u << 25));    // pad_amount : 4 DWORDs
  g1[1] = (int)((tensor_d0 & 0xFFFFu) << 16);                     // dim0[15:0]
  g1[2] = (int)((tensor_d0 >> 16) | ((tensor_d1 & 0xFFFFu) << 16)); // dim0[31:16], dim1[15:0]
  g1[3] = (int)((tensor_d1 >> 16) | (tile_d0 << 16));             // dim1[31:16], tile_dim0
  g1[4] = (int)(tile_d1 & 0xFFFFu);                               // tile_dim1 (tile_dim2=0)
  g1[5] = (int)stride0_elems;                                     // dim0_stride[31:0]
  g1[6] = 0;
  g1[7] = 0;
  i32x4 z4 = {0, 0, 0, 0};
#if TDM_SIX_ARGS
  i32x8 z8 = {0, 0, 0, 0, 0, 0, 0, 0};
  __builtin_amdgcn_tensor_load_to_lds(g0, g1, z4, z4, z8, 0);
#else
  __builtin_amdgcn_tensor_load_to_lds(g0, g1, z4, z4, 0);
#endif
#else
  (void)lds_byte_addr; (void)gsrc; (void)tensor_d0; (void)tensor_d1;
  (void)tile_d0; (void)tile_d1; (void)stride0_elems;
#endif
}

// ---------------------------------------------------------------------------
// Kernel 0: convert + transpose 5 weight matrices (512x512 f32 [K][N]) ->
//           bf16 [N][K] so GEMM B tiles are contiguous along K.
// ---------------------------------------------------------------------------
__global__ __launch_bounds__(256) void wconv_kernel(
    const float* __restrict__ Wq, const float* __restrict__ Wk,
    const float* __restrict__ Wv, const float* __restrict__ W1,
    const float* __restrict__ W2, __bf16* __restrict__ Wt) {
  const int which = blockIdx.y;
  const float* W = (which == 0) ? Wq : (which == 1) ? Wk : (which == 2) ? Wv
                 : (which == 3) ? W1 : W2;
  __bf16* o = Wt + (size_t)which * 512 * 512;
  int idx = blockIdx.x * 256 + threadIdx.x;   // idx = k*512 + n
  int k = idx >> 9, n = idx & 511;
  o[n * 512 + k] = (__bf16)W[idx];
}

// ---------------------------------------------------------------------------
// Kernel 1: LayerNorm (per-position gamma/beta) + bf16 casts + padding masks.
// ---------------------------------------------------------------------------
__global__ __launch_bounds__(256) void ln_kernel(
    const float* __restrict__ x, const float* __restrict__ gamma,
    const float* __restrict__ beta, __bf16* __restrict__ xb,
    __bf16* __restrict__ xnb, float* __restrict__ keym, float* __restrict__ qm) {
  const int row = blockIdx.x * 8 + (threadIdx.x >> 5);   // b*L + l
  const int lane = threadIdx.x & 31;
  const int l = row & 1023;
  const float* xr = x + (size_t)row * 512;

  float vals[16];
  float s = 0.f, ss = 0.f;
#pragma unroll
  for (int i = 0; i < 16; ++i) {
    float v = xr[lane + i * 32];
    vals[i] = v; s += v; ss += v * v;
  }
#pragma unroll
  for (int m = 1; m < 32; m <<= 1) {
    s  += __shfl_xor(s,  m, 32);
    ss += __shfl_xor(ss, m, 32);
  }
  const float mean = s * (1.f / 512.f);
  float var = ss * (1.f / 512.f) - mean * mean;
  const float rstd = rsqrtf(var + 1e-8f);

  float sn = 0.f;
#pragma unroll
  for (int i = 0; i < 16; ++i) {
    int j = lane + i * 32;
    float xn = gamma[l * 512 + j] * ((vals[i] - mean) * rstd) + beta[l * 512 + j];
    sn += xn;
    xnb[(size_t)row * 512 + j] = (__bf16)xn;
    xb [(size_t)row * 512 + j] = (__bf16)vals[i];
  }
#pragma unroll
  for (int m = 1; m < 32; m <<= 1) sn += __shfl_xor(sn, m, 32);
  if (lane == 0) {
    keym[row] = (s  != 0.f) ? 1.f : 0.f;   // sign(|sum(x)|)
    qm  [row] = (sn != 0.f) ? 1.f : 0.f;   // sign(|sum(x_norm)|)
  }
}

// ---------------------------------------------------------------------------
// GEMM tile machinery (128x128 tile / WG, 8 waves 4(M)x2(N), 32-wide K steps)
// ---------------------------------------------------------------------------
static __device__ inline void stage_tile_manual(__bf16* dst, const __bf16* __restrict__ src,
                                                int r0, int ks, int t) {
#pragma unroll
  for (int it = 0; it < 2; ++it) {
    int id = t + it * 256;          // 512 chunks of 16B = 128 rows x 4 segs
    int row = id >> 2, seg = id & 3;
    *(v8bf*)(dst + row * GST + seg * 8) =
        *(const v8bf*)(src + (size_t)(r0 + row) * 512 + ks * 32 + seg * 8);
  }
}

// Per-wave fragment compute for one 32-wide K step: acc[2][4] += A*B
static __device__ inline void tile_mma(const __bf16* As, const __bf16* Bs,
                                       int wm, int wn, int lane, v8f acc[2][4]) {
  const int ln = lane & 15, lg = lane >> 4;
  const int kb = lg * 16;     // A frag K base within 32-chunk
  const int kr = lg * 16;     // B frag K base within 32-chunk
  v16bf bfrag[4];
#pragma unroll
  for (int ni = 0; ni < 4; ++ni) {
    int c = wn * 64 + ni * 16 + ln;
    bfrag[ni] = ld16x2(Bs + c * GST + kr, Bs + c * GST + kr + 8);
  }
#pragma unroll
  for (int mi = 0; mi < 2; ++mi) {
    int r = wm * 32 + mi * 16 + ln;
    v16bf a = ld16x2(As + r * GST + kb, As + r * GST + kb + 8);
#pragma unroll
    for (int ni = 0; ni < 4; ++ni)
      acc[mi][ni] = WMMA_BF16(a, bfrag[ni], acc[mi][ni]);
  }
}

// Full K loop. TDM path: double-buffered async tensor loads issued by wave 0,
// synced with s_wait_tensorcnt + workgroup barrier. Fallback: manual staging.
static __device__ inline void gemm_loop(const __bf16* __restrict__ A,
                                        const __bf16* __restrict__ W,
                                        __bf16* As, __bf16* Bs,
                                        int m0, int n0, int t, int lane,
                                        int wm, int wn, unsigned Mrows,
                                        v8f acc[2][4]) {
#if USE_TDM
  const unsigned asb = (unsigned)(uintptr_t)As;
  const unsigned bsb = (unsigned)(uintptr_t)Bs;
  const unsigned bufb = TILE_HALVES * 2;   // bytes per tile buffer
  if (t < 32) {
    tdm_load_tile_2d(asb, A + (size_t)m0 * 512, 512, Mrows, 32, 128, 512);
    tdm_load_tile_2d(bsb, W + (size_t)n0 * 512, 512, 512,   32, 128, 512);
  }
  for (int ks = 0; ks < 16; ++ks) {
    const int cur = ks & 1, nxt = cur ^ 1;
    if (t < 32) {
      if (ks + 1 < 16) {
        tdm_load_tile_2d(asb + nxt * bufb, A + (size_t)m0 * 512 + (ks + 1) * 32,
                         512, Mrows, 32, 128, 512);
        tdm_load_tile_2d(bsb + nxt * bufb, W + (size_t)n0 * 512 + (ks + 1) * 32,
                         512, 512, 32, 128, 512);
        __builtin_amdgcn_s_wait_tensorcnt(2);   // current pair complete
      } else {
        __builtin_amdgcn_s_wait_tensorcnt(0);
      }
    }
    __syncthreads();
    tile_mma(As + cur * TILE_HALVES, Bs + cur * TILE_HALVES, wm, wn, lane, acc);
    __syncthreads();
  }
#else
  for (int ks = 0; ks < 16; ++ks) {
    stage_tile_manual(As, A, m0, ks, t);
    stage_tile_manual(Bs, W, n0, ks, t);
    __syncthreads();
    tile_mma(As, Bs, wm, wn, lane, acc);
    __syncthreads();
  }
#endif
  (void)Mrows;
}

#define GEMM_PROLOGUE()                                              \
  __shared__ __align__(16) __bf16 As[2 * TILE_HALVES];               \
  __shared__ __align__(16) __bf16 Bs[2 * TILE_HALVES];               \
  const int m0 = blockIdx.x * 128, n0 = blockIdx.y * 128;            \
  const int t = threadIdx.x, lane = t & 31, wv = t >> 5;             \
  const int wm = wv & 3, wn = wv >> 2;                               \
  const int ln = lane & 15, lg = lane >> 4;                          \
  v8f acc[2][4];                                                     \
  {                                                                  \
    const v8f zf = {0.f, 0.f, 0.f, 0.f, 0.f, 0.f, 0.f, 0.f};         \
    for (int i = 0; i < 2; ++i)                                      \
      for (int j = 0; j < 4; ++j) acc[i][j] = zf;                    \
  }

// ---------------------------------------------------------------------------
// Kernel 2: fused Q/K/V GEMM. blockIdx.z selects matrix. Output bf16 in
// (B*H, L, 64) head-major layout; Q pre-scaled by 1/sqrt(64).
// ---------------------------------------------------------------------------
__global__ __launch_bounds__(256) void qkv_gemm_kernel(
    const __bf16* __restrict__ xnb, const __bf16* __restrict__ xb,
    const __bf16* __restrict__ Wt,
    const float* __restrict__ bq, const float* __restrict__ bk,
    const float* __restrict__ bv,
    __bf16* __restrict__ qout, __bf16* __restrict__ kout, __bf16* __restrict__ vout) {
  GEMM_PROLOGUE()
  const int which = blockIdx.z;
  const __bf16* A = (which == 0) ? xnb : xb;
  const __bf16* W = Wt + (size_t)which * 512 * 512;
  const float* bias = (which == 0) ? bq : (which == 1) ? bk : bv;
  __bf16* outp = (which == 0) ? qout : (which == 1) ? kout : vout;
  const float scale = (which == 0) ? 0.125f : 1.0f;

  gemm_loop(A, W, As, Bs, m0, n0, t, lane, wm, wn, 16384u, acc);

#pragma unroll
  for (int mi = 0; mi < 2; ++mi)
#pragma unroll
    for (int ni = 0; ni < 4; ++ni) {
      int col = n0 + wn * 64 + ni * 16 + ln;
      int h = col >> 6, d = col & 63;
      float bb = bias[col];
#pragma unroll
      for (int r = 0; r < 8; ++r) {
        int m = m0 + wm * 32 + mi * 16 + r + 8 * lg;
        int b = m >> 10, l = m & 1023;
        float v = (acc[mi][ni][r] + bb) * scale;
        outp[((size_t)(b * 8 + h) * 1024 + l) * 64 + d] = (__bf16)v;
      }
    }
}

// ---------------------------------------------------------------------------
// Kernel 3: causal flash attention per (b,h). Each WG = 128 queries; each
// wave = 16 queries, streams 32-key blocks. S via 2 WMMAs/16-key tile;
// P->A-frag relayout through per-wave LDS; O += P*V via 4 WMMAs/block.
// Writes attn_out in f32 (residual for FFN2) and bf16 (A-matrix for FFN1).
// ---------------------------------------------------------------------------
__global__ __launch_bounds__(256) void attn_kernel(
    const __bf16* __restrict__ q, const __bf16* __restrict__ k,
    const __bf16* __restrict__ v, const float* __restrict__ keym,
    const float* __restrict__ qmsk, const __bf16* __restrict__ xnb,
    float* __restrict__ attn_out, __bf16* __restrict__ attn_out_bf) {
  __shared__ __align__(16) __bf16 Ps[8 * 16 * GST];

  const int bh = blockIdx.y, b = bh >> 3, h = bh & 7;
  const int w = threadIdx.x >> 5, lane = threadIdx.x & 31;
  const int ln = lane & 15, lg = lane >> 4;
  const int q0 = blockIdx.x * 128 + w * 16;     // first query row of this wave
  const int qlast = q0 + 15;

  const __bf16* qptr = q + (size_t)bh * 1024 * 64;
  const __bf16* kptr = k + (size_t)bh * 1024 * 64;
  const __bf16* vptr = v + (size_t)bh * 1024 * 64;
  __bf16* P = Ps + w * 16 * GST;

  // Q A-fragments for d=0..31 and d=32..63 (already scaled by 1/8)
  const int qrow = q0 + ln;
  const int kb8 = lg * 16;
  const __bf16* qr = qptr + (size_t)qrow * 64;
  v16bf qf0 = ld16x2(qr + kb8,      qr + kb8 + 8);
  v16bf qf1 = ld16x2(qr + 32 + kb8, qr + 32 + kb8 + 8);

  float mrun[8], lsum[8];
  v8f o[4];
  const v8f zf = {0.f, 0.f, 0.f, 0.f, 0.f, 0.f, 0.f, 0.f};
#pragma unroll
  for (int r = 0; r < 8; ++r) { mrun[r] = -3.0e38f; lsum[r] = 0.f; }
#pragma unroll
  for (int tt = 0; tt < 4; ++tt) o[tt] = zf;

  const int rowg_base = q0 + 8 * lg;

  for (int kb = 0; kb <= qlast; kb += 32) {
    // ---- S = Q * K^T for 32 keys (two 16-key column tiles) ----
    int key0 = kb + ln;      if (key0 > qlast) key0 = qlast;   // clamped; masked below
    int key1 = kb + 16 + ln; if (key1 > qlast) key1 = qlast;
    const int dr = lg * 16;
    const __bf16* k0p = kptr + (size_t)key0 * 64;
    const __bf16* k1p = kptr + (size_t)key1 * 64;
    v16bf kf00 = ld16x2(k0p + dr,      k0p + dr + 8);
    v16bf kf01 = ld16x2(k0p + 32 + dr, k0p + 32 + dr + 8);
    v16bf kf10 = ld16x2(k1p + dr,      k1p + dr + 8);
    v16bf kf11 = ld16x2(k1p + 32 + dr, k1p + 32 + dr + 8);
    v8f s0 = zf, s1 = zf;
    s0 = WMMA_BF16(qf0, kf00, s0);
    s0 = WMMA_BF16(qf1, kf01, s0);
    s1 = WMMA_BF16(qf0, kf10, s1);
    s1 = WMMA_BF16(qf1, kf11, s1);

    // ---- masks + online softmax (rows live across 16-lane half-waves) ----
    const int col0 = kb + ln, col1 = kb + 16 + ln;
    const float km0 = (col0 <= qlast) ? keym[(size_t)b * 1024 + col0] : 0.f;
    const float km1 = (col1 <= qlast) ? keym[(size_t)b * 1024 + col1] : 0.f;
#pragma unroll
    for (int r = 0; r < 8; ++r) {
      const int rowg = rowg_base + r;
      float a0 = s0[r], a1 = s1[r];
      if (col0 > rowg || km0 == 0.f) a0 = NEGV;
      if (col1 > rowg || km1 == 0.f) a1 = NEGV;
      float bm = fmaxf(a0, a1);
      bm = fmaxf(bm, __shfl_xor(bm, 1, 32));
      bm = fmaxf(bm, __shfl_xor(bm, 2, 32));
      bm = fmaxf(bm, __shfl_xor(bm, 4, 32));
      bm = fmaxf(bm, __shfl_xor(bm, 8, 32));
      const float nm = fmaxf(mrun[r], bm);
      const float sc = __expf(mrun[r] - nm);
      const float p0 = __expf(a0 - nm);
      const float p1 = __expf(a1 - nm);
      float bs = p0 + p1;
      bs += __shfl_xor(bs, 1, 32);
      bs += __shfl_xor(bs, 2, 32);
      bs += __shfl_xor(bs, 4, 32);
      bs += __shfl_xor(bs, 8, 32);
      lsum[r] = lsum[r] * sc + bs;
      mrun[r] = nm;
#pragma unroll
      for (int tt = 0; tt < 4; ++tt) o[tt][r] *= sc;
      P[(r + 8 * lg) * GST + ln]      = (__bf16)p0;
      P[(r + 8 * lg) * GST + 16 + ln] = (__bf16)p1;
    }

    // ---- O += P * V : relayout P through LDS into A-frag, V as B-frags ----
    v16bf pf = ld16x2(P + ln * GST + kb8, P + ln * GST + kb8 + 8);
#pragma unroll
    for (int tt = 0; tt < 4; ++tt) {
      const int dcol = tt * 16 + ln;
      v16bf vf;
#pragma unroll
      for (int j = 0; j < 8; ++j) {
        int ki0 = kb + lg * 16 + 2 * j;
        int ki1 = ki0 + 1;
        if (ki0 > qlast) ki0 = qlast;     // P is ~0 there
        if (ki1 > qlast) ki1 = qlast;
        vf[2 * j]     = vptr[(size_t)ki0 * 64 + dcol];
        vf[2 * j + 1] = vptr[(size_t)ki1 * 64 + dcol];
      }
      o[tt] = WMMA_BF16(pf, vf, o[tt]);
    }
  }

  // ---- epilogue: 1/lsum, query mask, residual with x_norm ----
#pragma unroll
  for (int r = 0; r < 8; ++r) {
    const int rowg = rowg_base + r;
    const float inv = (lsum[r] > 0.f) ? (1.f / lsum[r]) : 0.f;
    const float f = inv * qmsk[(size_t)b * 1024 + rowg];
    const size_t base = ((size_t)b * 1024 + rowg) * 512 + h * 64;
#pragma unroll
    for (int tt = 0; tt < 4; ++tt) {
      const int dcol = tt * 16 + ln;
      const float val = o[tt][r] * f + (float)xnb[base + dcol];
      attn_out[base + dcol]    = val;
      attn_out_bf[base + dcol] = (__bf16)val;
    }
  }
}

// ---------------------------------------------------------------------------
// Kernel 4: FFN layer 1 — h = relu(attn_out @ W1 + b1), bf16 out.
// ---------------------------------------------------------------------------
__global__ __launch_bounds__(256) void ffn1_kernel(
    const __bf16* __restrict__ attn_bf, const __bf16* __restrict__ W1t,
    const float* __restrict__ b1, __bf16* __restrict__ hout) {
  GEMM_PROLOGUE()
  gemm_loop(attn_bf, W1t, As, Bs, m0, n0, t, lane, wm, wn, 16384u, acc);

#pragma unroll
  for (int mi = 0; mi < 2; ++mi)
#pragma unroll
    for (int ni = 0; ni < 4; ++ni) {
      int col = n0 + wn * 64 + ni * 16 + ln;
      float bb = b1[col];
#pragma unroll
      for (int r = 0; r < 8; ++r) {
        int m = m0 + wm * 32 + mi * 16 + r + 8 * lg;
        float v = acc[mi][ni][r] + bb;
        v = (v > 0.f) ? v : 0.f;
        hout[(size_t)m * 512 + col] = (__bf16)v;
      }
    }
}

// ---------------------------------------------------------------------------
// Kernel 5: FFN layer 2 — out = (h @ W2 + b2 + attn_out) * mask, f32 out.
// ---------------------------------------------------------------------------
__global__ __launch_bounds__(256) void ffn2_kernel(
    const __bf16* __restrict__ hbuf, const __bf16* __restrict__ W2t,
    const float* __restrict__ b2, const float* __restrict__ attn,
    const float* __restrict__ mask, float* __restrict__ out) {
  GEMM_PROLOGUE()
  gemm_loop(hbuf, W2t, As, Bs, m0, n0, t, lane, wm, wn, 16384u, acc);

#pragma unroll
  for (int mi = 0; mi < 2; ++mi)
#pragma unroll
    for (int ni = 0; ni < 4; ++ni) {
      int col = n0 + wn * 64 + ni * 16 + ln;
      float bb = b2[col];
#pragma unroll
      for (int r = 0; r < 8; ++r) {
        int m = m0 + wm * 32 + mi * 16 + r + 8 * lg;
        float v = acc[mi][ni][r] + bb + attn[(size_t)m * 512 + col];
        out[(size_t)m * 512 + col] = v * mask[m];
      }
    }
}

// ---------------------------------------------------------------------------
// Host launch
// ---------------------------------------------------------------------------
extern "C" void kernel_launch(void* const* d_in, const int* in_sizes, int n_in,
                              void* d_out, int out_size, void* d_ws, size_t ws_size,
                              hipStream_t stream) {
  (void)in_sizes; (void)n_in; (void)out_size; (void)ws_size;
  const float* x     = (const float*)d_in[0];
  const float* mask  = (const float*)d_in[1];
  const float* gamma = (const float*)d_in[2];
  const float* beta  = (const float*)d_in[3];
  const float* Wq = (const float*)d_in[4];
  const float* bq = (const float*)d_in[5];
  const float* Wk = (const float*)d_in[6];
  const float* bk = (const float*)d_in[7];
  const float* Wv = (const float*)d_in[8];
  const float* bv = (const float*)d_in[9];
  const float* W1 = (const float*)d_in[10];
  const float* b1 = (const float*)d_in[11];
  const float* W2 = (const float*)d_in[12];
  const float* b2 = (const float*)d_in[13];
  float* out = (float*)d_out;

  char* ws = (char*)d_ws;
  size_t off = 0;
  auto alloc = [&](size_t bytes) -> void* {
    void* p = ws + off;
    off += (bytes + 255) & ~(size_t)255;
    return p;
  };
  const size_t M = 16384, Dd = 512;
  __bf16* xb   = (__bf16*)alloc(M * Dd * 2);
  __bf16* xnb  = (__bf16*)alloc(M * Dd * 2);
  __bf16* Wt   = (__bf16*)alloc(5 * Dd * Dd * 2);
  __bf16* qb   = (__bf16*)alloc(M * Dd * 2);
  __bf16* kbuf = (__bf16*)alloc(M * Dd * 2);
  __bf16* vbuf = (__bf16*)alloc(M * Dd * 2);
  float*  ao   = (float*)alloc(M * Dd * 4);
  __bf16* aob  = (__bf16*)alloc(M * Dd * 2);
  __bf16* hb   = (__bf16*)alloc(M * Dd * 2);
  float*  keym = (float*)alloc(M * 4);
  float*  qmm  = (float*)alloc(M * 4);

  wconv_kernel<<<dim3(1024, 5), 256, 0, stream>>>(Wq, Wk, Wv, W1, W2, Wt);
  ln_kernel<<<2048, 256, 0, stream>>>(x, gamma, beta, xb, xnb, keym, qmm);
  qkv_gemm_kernel<<<dim3(128, 4, 3), 256, 0, stream>>>(xnb, xb, Wt, bq, bk, bv,
                                                       qb, kbuf, vbuf);
  attn_kernel<<<dim3(8, 128), 256, 0, stream>>>(qb, kbuf, vbuf, keym, qmm, xnb,
                                                ao, aob);
  ffn1_kernel<<<dim3(128, 4), 256, 0, stream>>>(aob, Wt + (size_t)3 * 512 * 512,
                                                b1, hb);
  ffn2_kernel<<<dim3(128, 4), 256, 0, stream>>>(hb, Wt + (size_t)4 * 512 * 512, b2,
                                                ao, mask, out);
}
